// GanDTI_13262859010097
// MI455X (gfx1250) — matrix operand order repaired and checked by reading, not compiled, not run on hardware
//
#include <hip/hip_runtime.h>
#include <hip/hip_bf16.h>

typedef __attribute__((ext_vector_type(16))) _Float16 v16h;
typedef __attribute__((ext_vector_type(8)))  _Float16 v8h;
typedef __attribute__((ext_vector_type(8)))  float    v8f;

#define NB   4096
#define NA   50
#define NF   40
#define K30  30
#define N1001 1001
#define KP_A69 1024   // 1001 padded to mult of 32
#define KP_PROT 512
#define KP_GNN 64     // 40 padded
#define NP48 48       // 40 padded to mult of 16

// ---------------------------------------------------------------------------
// f32 -> f16 convert, row-major with zero padding (activation staging)
// ---------------------------------------------------------------------------
__global__ void __launch_bounds__(256) k_cvt(const float* __restrict__ src,
                                             _Float16* __restrict__ dst,
                                             int srcRows, int srcCols,
                                             int dstRows, int dstCols) {
    int idx = blockIdx.x * blockDim.x + threadIdx.x;
    if (idx >= dstRows * dstCols) return;
    int c = idx % dstCols, r = idx / dstCols;
    dst[idx] = (r < srcRows && c < srcCols) ? (_Float16)src[(size_t)r * srcCols + c]
                                            : (_Float16)0.f;
}

// ---------------------------------------------------------------------------
// f32 [K,N] -> f16 transposed BT [Npad][Kpad], zero padded.
// Makes the WMMA B-fragment a contiguous 32B (v16h) load per lane.
// ---------------------------------------------------------------------------
__global__ void __launch_bounds__(256) k_cvt_t(const float* __restrict__ src,
                                               _Float16* __restrict__ dstT,
                                               int srcK, int srcN,
                                               int Kp, int Np) {
    int idx = blockIdx.x * blockDim.x + threadIdx.x;
    if (idx >= Np * Kp) return;
    int k = idx % Kp, n = idx / Kp;
    dstT[idx] = (k < srcK && n < srcN) ? (_Float16)src[(size_t)k * srcN + n]
                                       : (_Float16)0.f;
}

// ---------------------------------------------------------------------------
// a69[b,i] = A69[b,i,:] . W1 + b1   (bandwidth-dominant: 492 MB stream)
// emits f16, K-padded to 1024 columns for the following WMMA GEMM
// ---------------------------------------------------------------------------
__global__ void __launch_bounds__(256) k_a69(const float* __restrict__ A69,
                                             const float* __restrict__ W1,
                                             const float* __restrict__ b1,
                                             _Float16* __restrict__ a69h) {
    int idx = blockIdx.x * blockDim.x + threadIdx.x;   // NB * 1024
    if (idx >= NB * KP_A69) return;
    int i = idx & (KP_A69 - 1), b = idx >> 10;
    if (i < N1001) {
        const float* row = A69 + ((size_t)b * N1001 + i) * K30;
        float s = b1[0];
        #pragma unroll
        for (int k = 0; k < K30; ++k) s += row[k] * W1[k];
        a69h[idx] = (_Float16)s;
    } else {
        a69h[idx] = (_Float16)0.f;
    }
}

// ---------------------------------------------------------------------------
// Embedding gather: comp[b,i,f] = emb[atoms[b,i], f]; also f16 K-padded copy
// ---------------------------------------------------------------------------
__global__ void __launch_bounds__(256) k_embed(const int* __restrict__ atoms,
                                               const float* __restrict__ emb,
                                               float* __restrict__ comp,
                                               _Float16* __restrict__ comph) {
    int idx = blockIdx.x * blockDim.x + threadIdx.x;   // NB*NA*64
    if (idx >= NB * NA * KP_GNN) return;
    int f = idx & (KP_GNN - 1), r = idx >> 6;          // r = b*NA + i
    if (f < NF) {
        float v = emb[(size_t)atoms[r] * NF + f];
        comp[(size_t)r * NF + f] = v;
        comph[idx] = (_Float16)v;
    } else {
        comph[idx] = (_Float16)0.f;
    }
}

// ---------------------------------------------------------------------------
// Wave-tile WMMA GEMM: C[M, nStore] = act(A[M,Kp] * B + bias)
// B given pre-transposed/padded as BT[Npad][Kp].
// One wave computes NT adjacent 16x16 C tiles (A fragment reused NT times).
// Fragment layouts used here (wave32, contiguous k order for both A and B):
//   A: lane(l16)=row, half selects K group of 16 {half*16+j}
//   B: lane(l16)=col, half selects K group of 16 -> contiguous in BT
//   C: v8f, M = m0 + half*8 + r, N = n0 + l16
// ACT: 0=none, 1=relu, 2=leaky(0.01)
// ---------------------------------------------------------------------------
template<int NT, int ACT>
__global__ void __launch_bounds__(256) k_gemm_wmma(
    const _Float16* __restrict__ A, const _Float16* __restrict__ BT,
    const float* __restrict__ bias, float* __restrict__ C,
    int M, int Kp, int nGroups, int ldc, int nStore) {

    int wave  = (blockIdx.x * blockDim.x + threadIdx.x) >> 5;
    int lane  = threadIdx.x & 31;
    int mTile = wave / nGroups;
    int g     = wave - mTile * nGroups;
    int m0 = mTile * 16;
    if (m0 >= M) return;                 // uniform per wave -> EXEC stays all-1s
    int half = lane >> 4, l16 = lane & 15;

    const _Float16* arow = A + (size_t)(m0 + l16) * Kp;
    const _Float16* brow[NT];
    #pragma unroll
    for (int t = 0; t < NT; ++t)
        brow[t] = BT + (size_t)(g * (NT * 16) + t * 16 + l16) * Kp + half * 16;

    v8f acc[NT];
    #pragma unroll
    for (int t = 0; t < NT; ++t) acc[t] = (v8f){};

    for (int k0 = 0; k0 < Kp; k0 += 32) {
        // A fragment: two contiguous 16B chunks -> 2x global_load_b128
        v8h alo = *(const v8h*)(arow + k0 + half * 16);
        v8h ahi = *(const v8h*)(arow + k0 + half * 16 + 8);
        v16h af;
        #pragma unroll
        for (int j = 0; j < 8; ++j) { af[j] = alo[j]; af[8 + j] = ahi[j]; }

        #pragma unroll
        for (int t = 0; t < NT; ++t) {
            // B fragment: one contiguous 32B chunk (32B-aligned) per lane
            v16h bf = *(const v16h*)(brow[t] + k0);
            acc[t] = __builtin_amdgcn_wmma_f32_16x16x32_f16(
                false, af, false, bf, (short)0, acc[t], false, false);
        }
    }

    #pragma unroll
    for (int t = 0; t < NT; ++t) {
        int n = g * (NT * 16) + t * 16 + l16;
        if (n < nStore) {
            float bv = bias[n];
            #pragma unroll
            for (int r = 0; r < 8; ++r) {
                float v = acc[t][r] + bv;
                if (ACT == 1)      v = fmaxf(v, 0.f);
                else if (ACT == 2) v = v > 0.f ? v : 0.01f * v;
                C[(size_t)(m0 + half * 8 + r) * ldc + n] = v;
            }
        }
    }
}

// ---------------------------------------------------------------------------
// comp += A_adj @ h  (batched dense 50x50 * 50x40); refresh f16 copy of comp
// ---------------------------------------------------------------------------
__global__ void __launch_bounds__(256) k_spmm(const float* __restrict__ Aadj,
                                              const float* __restrict__ h,
                                              float* __restrict__ comp,
                                              _Float16* __restrict__ comph) {
    int idx = blockIdx.x * blockDim.x + threadIdx.x;   // NB*NA*NF
    if (idx >= NB * NA * NF) return;
    int f = idx % NF; int t = idx / NF; int i = t % NA; int b = t / NA;
    const float* arow = Aadj + ((size_t)b * NA + i) * NA;
    const float* hb   = h    + (size_t)b * NA * NF + f;
    float acc = comp[idx];
    #pragma unroll 5
    for (int j = 0; j < NA; ++j) acc += arow[j] * hb[(size_t)j * NF];
    comp[idx] = acc;
    comph[((size_t)b * NA + i) * KP_GNN + f] = (_Float16)acc;
}

// ---------------------------------------------------------------------------
// compound_vec[b,f] = mean_i( comp[b,i,f] + residual emb[atoms[b,i],f] )
// ---------------------------------------------------------------------------
__global__ void __launch_bounds__(256) k_mean(const int* __restrict__ atoms,
                                              const float* __restrict__ emb,
                                              const float* __restrict__ comp,
                                              float* __restrict__ cvec) {
    int idx = blockIdx.x * blockDim.x + threadIdx.x;   // NB*NF
    if (idx >= NB * NF) return;
    int f = idx % NF, b = idx / NF;
    const float* cb = comp + (size_t)b * NA * NF + f;
    const int* ab = atoms + (size_t)b * NA;
    float s = 0.f;
    #pragma unroll 5
    for (int i = 0; i < NA; ++i) s += cb[(size_t)i * NF] + emb[(size_t)ab[i] * NF + f];
    cvec[idx] = s * (1.f / NA);
}

// ---------------------------------------------------------------------------
// pcat = concat(a256, p256) -> f16 for the Wp GEMM
// ---------------------------------------------------------------------------
__global__ void __launch_bounds__(256) k_pcat(const float* __restrict__ a256,
                                              const float* __restrict__ p256,
                                              _Float16* __restrict__ pcath) {
    int idx = blockIdx.x * blockDim.x + threadIdx.x;   // NB*512
    if (idx >= NB * 512) return;
    int j = idx & 511, b = idx >> 9;
    float v = (j < 256) ? a256[(size_t)b * 256 + j] : p256[(size_t)b * 256 + j - 256];
    pcath[idx] = (_Float16)v;
}

// ---------------------------------------------------------------------------
// Fused head per sample: tanh attention gate + 2-layer MLP + output dot
// ---------------------------------------------------------------------------
__global__ void __launch_bounds__(128) k_head(const float* __restrict__ cvec,
                                              const float* __restrict__ p40,
                                              const float* __restrict__ Watt,
                                              const float* __restrict__ batt,
                                              const float* __restrict__ Wm,
                                              const float* __restrict__ bm,
                                              const float* __restrict__ Wo,
                                              const float* __restrict__ bo,
                                              float* __restrict__ out) {
    __shared__ float pr[NF], cv[NF], ph[NF], cp[2 * NF], red[2 * NF];
    int b = blockIdx.x, t = threadIdx.x;
    if (t < NF) { pr[t] = p40[(size_t)b * NF + t]; cv[t] = cvec[(size_t)b * NF + t]; }
    __syncthreads();
    if (t < NF) {
        float s = batt[t];
        #pragma unroll 8
        for (int j = 0; j < NF; ++j) s += pr[j] * Watt[(size_t)j * NF + t];
        float r = fmaxf(s, 0.f);
        ph[t] = r;
        red[t] = cv[t] * r;
    }
    __syncthreads();
    if (t == 0) {
        float m = 0.f;
        for (int j = 0; j < NF; ++j) m += red[j];
        red[0] = tanhf(m);
    }
    __syncthreads();
    float w = red[0];
    if (t < NF) { cp[t] = cv[t]; cp[NF + t] = w * ph[t]; }
    __syncthreads();
    for (int L = 0; L < 2; ++L) {
        float s = 0.f;
        if (t < 2 * NF) {
            s = bm[(size_t)L * 2 * NF + t];
            #pragma unroll 8
            for (int j = 0; j < 2 * NF; ++j)
                s += cp[j] * Wm[((size_t)L * 2 * NF + j) * 2 * NF + t];
            s = fmaxf(s, 0.f);
        }
        __syncthreads();
        if (t < 2 * NF) cp[t] = s;
        __syncthreads();
    }
    if (t < 2 * NF) red[t] = cp[t] * Wo[t];
    __syncthreads();
    if (t == 0) {
        float s = bo[0];
        for (int j = 0; j < 2 * NF; ++j) s += red[j];
        out[b] = s;
    }
}

// ---------------------------------------------------------------------------
extern "C" void kernel_launch(void* const* d_in, const int* in_sizes, int n_in,
                              void* d_out, int out_size, void* d_ws, size_t ws_size,
                              hipStream_t stream) {
    (void)in_sizes; (void)n_in; (void)out_size; (void)ws_size;
    const int*   atoms   = (const int*)  d_in[0];
    const float* Aadj    = (const float*)d_in[1];
    const float* A69     = (const float*)d_in[2];
    const float* protein = (const float*)d_in[3];
    const float* emb     = (const float*)d_in[4];
    const float* Wg      = (const float*)d_in[5];
    const float* bg      = (const float*)d_in[6];
    const float* Watt    = (const float*)d_in[7];
    const float* batt    = (const float*)d_in[8];
    const float* W1      = (const float*)d_in[9];
    const float* b1      = (const float*)d_in[10];
    const float* W2      = (const float*)d_in[11];
    const float* b2      = (const float*)d_in[12];
    const float* W3      = (const float*)d_in[13];
    const float* b3      = (const float*)d_in[14];
    const float* Wp      = (const float*)d_in[15];
    const float* bp      = (const float*)d_in[16];
    const float* Wm      = (const float*)d_in[17];
    const float* bm      = (const float*)d_in[18];
    const float* Wo      = (const float*)d_in[19];
    const float* bo      = (const float*)d_in[20];
    float* out = (float*)d_out;

    // ---- workspace carve (256B aligned) ----
    char* ws = (char*)d_ws;
    size_t off = 0;
    auto carve = [&](size_t bytes) {
        char* p = ws + off;
        off = (off + bytes + 255) & ~(size_t)255;
        return p;
    };
    float*    a256  = (float*)   carve((size_t)NB * 256 * 4);
    float*    p256  = (float*)   carve((size_t)NB * 256 * 4);
    float*    p40   = (float*)   carve((size_t)NB * NF * 4);
    float*    comp  = (float*)   carve((size_t)NB * NA * NF * 4);
    float*    hbuf  = (float*)   carve((size_t)NB * NA * NF * 4);
    float*    cvec  = (float*)   carve((size_t)NB * NF * 4);
    _Float16* a69h  = (_Float16*)carve((size_t)NB * KP_A69 * 2);
    _Float16* W2T   = (_Float16*)carve((size_t)256 * KP_A69 * 2);   // [N=256][K=1024]
    _Float16* proth = (_Float16*)carve((size_t)NB * KP_PROT * 2);
    _Float16* W3T   = (_Float16*)carve((size_t)256 * KP_PROT * 2);  // [256][512]
    _Float16* pcath = (_Float16*)carve((size_t)NB * 512 * 2);
    _Float16* WpT   = (_Float16*)carve((size_t)NP48 * KP_PROT * 2); // [48][512]
    _Float16* comph = (_Float16*)carve((size_t)NB * NA * KP_GNN * 2);
    _Float16* WgT   = (_Float16*)carve((size_t)3 * NP48 * KP_GNN * 2); // 3x[48][64]

    auto blocks = [](long long n, int bs) { return (unsigned)((n + bs - 1) / bs); };

    // ---- weight staging: f16, transposed, zero padded ----
    k_cvt_t<<<blocks((long long)256 * KP_A69, 256), 256, 0, stream>>>(W2, W2T, N1001, 256, KP_A69, 256);
    k_cvt_t<<<blocks((long long)256 * KP_PROT, 256), 256, 0, stream>>>(W3, W3T, 512, 256, KP_PROT, 256);
    k_cvt_t<<<blocks((long long)NP48 * KP_PROT, 256), 256, 0, stream>>>(Wp, WpT, 512, NF, KP_PROT, NP48);
    for (int i = 0; i < 3; ++i)
        k_cvt_t<<<blocks((long long)NP48 * KP_GNN, 256), 256, 0, stream>>>(
            Wg + (size_t)i * NF * NF, WgT + (size_t)i * NP48 * KP_GNN, NF, NF, KP_GNN, NP48);
    // activation staging (row-major)
    k_cvt<<<blocks((long long)NB * 512, 256), 256, 0, stream>>>(protein, proth, NB, 512, NB, 512);

    // ---- A69 reduction (HBM-bound) + embedding gather ----
    k_a69<<<blocks((long long)NB * KP_A69, 256), 256, 0, stream>>>(A69, W1, b1, a69h);
    k_embed<<<blocks((long long)NB * NA * KP_GNN, 256), 256, 0, stream>>>(atoms, emb, comp, comph);

    // ---- WMMA GEMMs: a256 = a69 @ W2 + b2 ; p256 = protein @ W3 + b3 ----
    // M=4096 (256 mTiles), N=256 -> NT=4, nGroups=4 -> 1024 waves -> 128 blocks
    k_gemm_wmma<4, 0><<<128, 256, 0, stream>>>(a69h, W2T, b2, a256,
                                               NB, KP_A69, 4, 256, 256);
    k_gemm_wmma<4, 0><<<128, 256, 0, stream>>>(proth, W3T, b3, p256,
                                               NB, KP_PROT, 4, 256, 256);

    // ---- pcat -> p40 = pcat @ Wp + bp ----
    k_pcat<<<blocks((long long)NB * 512, 256), 256, 0, stream>>>(a256, p256, pcath);
    // M=4096, Npad=48 -> NT=3, nGroups=1 -> 256 waves -> 32 blocks
    k_gemm_wmma<3, 0><<<32, 256, 0, stream>>>(pcath, WpT, bp, p40,
                                              NB, KP_PROT, 1, NF, NF);

    // ---- GNN: 3 x { h = leaky(comp @ Wg_i + bg_i) ; comp += A @ h } ----
    for (int i = 0; i < 3; ++i) {
        // M=204800 (12800 mTiles), Npad=48 -> NT=3, nGroups=1 -> 12800 waves -> 1600 blocks
        k_gemm_wmma<3, 2><<<1600, 256, 0, stream>>>(comph, WgT + (size_t)i * NP48 * KP_GNN,
                                                    bg + (size_t)i * NF, hbuf,
                                                    NB * NA, KP_GNN, 1, NF, NF);
        k_spmm<<<blocks((long long)NB * NA * NF, 256), 256, 0, stream>>>(Aadj, hbuf, comp, comph);
    }

    // ---- mean + residual, then fused attention/MLP head ----
    k_mean<<<blocks((long long)NB * NF, 256), 256, 0, stream>>>(atoms, emb, comp, cvec);
    k_head<<<NB, 128, 0, stream>>>(cvec, p40, Watt, batt, Wm, bm, Wo, bo, out);
}
